// CfcCell_37546604101689
// MI455X (gfx1250) — compile-verified, matrix operand order checked by reading, not run on hardware
//
#include <hip/hip_runtime.h>

// ---------------------------------------------------------------------------
// CfC cell for MI455X (gfx1250): bf16 WMMA GEMM chain, fp32 accumulate.
//   M = B*T = 131072 rows
//   stage1: X=[input|hx] [M,768] @ W0[768,1024] -> lecun_tanh -> H1 bf16
//   stage2: H1 [M,1024] @ W1[1024,1024] -> lecun_tanh -> H2 bf16
//   heads : H2 @ {Wff1,Wff2,Wta,Wtb}[1024,512] -> gate -> out fp32 [M,512]
// Weights are converted fp32->bf16 AND transposed to [N,K] once per launch, so
// all GEMM staging is contiguous row copies (async global->LDS when available).
// ---------------------------------------------------------------------------

typedef __attribute__((ext_vector_type(16))) __bf16 v16bf;
typedef __attribute__((ext_vector_type(8)))  float  v8f;
typedef __attribute__((ext_vector_type(4)))  int    v4i;

#define M_TOTAL (128 * 1024)
#define PITCH   40            // bf16 elements per LDS tile row (80B, 16B-aligned chunks)

#if __has_builtin(__builtin_amdgcn_global_load_async_to_lds_b128)
#define HAVE_ASYNC 1
#else
#define HAVE_ASYNC 0
#endif

__device__ __forceinline__ float fast_tanh(float x) {
    float ax = __builtin_fabsf(x);
    float e  = __expf(-2.0f * ax);
    float t  = (1.0f - e) / (1.0f + e);
    return x < 0.0f ? -t : t;
}
__device__ __forceinline__ float lecun_act(float x) {
    return 1.7159f * fast_tanh(0.666f * x);
}
__device__ __forceinline__ float fast_sigmoid(float x) {
    return 1.0f / (1.0f + __expf(-x));
}

__device__ __forceinline__ void wait_async0() {
#if HAVE_ASYNC
#if __has_builtin(__builtin_amdgcn_s_wait_asynccnt)
    __builtin_amdgcn_s_wait_asynccnt(0);
#else
    asm volatile("s_wait_asynccnt 0x0" ::: "memory");
#endif
#endif
}

// Async copy of 16 bytes global -> LDS (one b128 transfer, ASYNCcnt-tracked).
__device__ __forceinline__ void async_cp16(const void* g, void* l) {
#if HAVE_ASYNC
    __attribute__((address_space(1))) v4i* gp =
        (__attribute__((address_space(1))) v4i*)(unsigned long long)(size_t)g;
    __attribute__((address_space(3))) v4i* lp =
        (__attribute__((address_space(3))) v4i*)(unsigned)(size_t)l;
    __builtin_amdgcn_global_load_async_to_lds_b128(gp, lp, 0, 0);
#else
    (void)g; (void)l;
#endif
}
// 32 bytes (one LDS tile row-chunk of 16 bf16)
__device__ __forceinline__ void async_cp32(const void* g, void* l) {
    async_cp16(g, l);
    async_cp16((const char*)g + 16, (char*)l + 16);
}

union FragU { unsigned u[8]; v16bf v; };

// Load one 16x32 bf16 fragment from an LDS tile stored [rowOrCol][k], pitch PITCH.
__device__ __forceinline__ v16bf load_frag(const __bf16* tile, int rc, int idx, int half) {
    const __bf16* p = tile + (rc + idx) * PITCH + half * 8;
    FragU f;
    f.u[0] = *(const unsigned*)(p + 0);
    f.u[1] = *(const unsigned*)(p + 2);
    f.u[2] = *(const unsigned*)(p + 4);
    f.u[3] = *(const unsigned*)(p + 6);
    f.u[4] = *(const unsigned*)(p + 16);
    f.u[5] = *(const unsigned*)(p + 18);
    f.u[6] = *(const unsigned*)(p + 20);
    f.u[7] = *(const unsigned*)(p + 22);
    return f.v;
}

// ---------------------------------------------------------------------------
// Weight convert + transpose: s[K,N] fp32 -> d[N,K] bf16. One-time cost.
// grid = (N/256, K); coalesced reads, scattered bf16 writes (weights are tiny).
// ---------------------------------------------------------------------------
__global__ __launch_bounds__(256) void cvt_transpose_bf16(const float* __restrict__ s,
                                                          __bf16* __restrict__ d,
                                                          int K, int N) {
    const int n = blockIdx.x * 256 + threadIdx.x;
    const int k = blockIdx.y;
    d[(size_t)n * K + k] = (__bf16)s[(size_t)k * N + n];
}

// ---------------------------------------------------------------------------
// Backbone GEMM: C[M,1024] = lecun_tanh(A @ W + b)
//   MODE 0: A = concat(input[M,256], hx[M,512]) fp32, K=768 (cvt bf16 on stage)
//   MODE 1: A = bf16 [M,1024], K=1024
// Wt = weights bf16 TRANSPOSED [1024, K].
// Workgroup tile 128x128, K-chunk 32, 8 waves, wave tile 64x32, double-buffered.
// ---------------------------------------------------------------------------
template <int MODE>
__global__ __launch_bounds__(256) void gemm_backbone(
    const float*  __restrict__ Ain,
    const float*  __restrict__ Ahx,
    const __bf16* __restrict__ Abf,
    const __bf16* __restrict__ Wt,    // [1024, K] bf16 (transposed)
    const float*  __restrict__ bias,  // [1024]
    __bf16*       __restrict__ Out)   // [M,1024]
{
    constexpr int K  = (MODE == 0) ? 768 : 1024;
    constexpr int NC = K / 32;

    __shared__ __bf16 As[2][128 * PITCH];
    __shared__ __bf16 Bs[2][128 * PITCH];

    const int tid  = threadIdx.x;
    const int lane = tid & 31;
    const int wid  = tid >> 5;
    const int idx  = lane & 15;
    const int half = lane >> 4;

    const int m0 = blockIdx.y * 128;
    const int n0 = blockIdx.x * 128;
    const int rowbase = (wid >> 2) * 64;  // 0 / 64
    const int colbase = (wid & 3) * 32;   // 0..96

    // staging geometry: each thread owns one 16-element row chunk of each tile
    const int arow  = tid >> 1;           // 0..127
    const int ahalf = (tid & 1) * 16;     // 0 / 16

    float4 fa[4];                          // MODE 0 A staging (fp32)
    uint4  ua[2];                          // fallback A staging (bf16)
    uint4  ub[2];                          // fallback B staging

    auto fetchA = [&](int kc) {
        const int k0 = kc * 32;
        if (MODE == 0) {
            const int gk = k0 + ahalf;
            const float* src = (gk < 256)
                ? (Ain + (size_t)(m0 + arow) * 256 + gk)
                : (Ahx + (size_t)(m0 + arow) * 512 + (gk - 256));
            const float4* p4 = (const float4*)src;
            fa[0] = p4[0]; fa[1] = p4[1]; fa[2] = p4[2]; fa[3] = p4[3];
        } else {
            const uint4* p4 = (const uint4*)(Abf + (size_t)(m0 + arow) * 1024 + k0 + ahalf);
            ua[0] = p4[0]; ua[1] = p4[1];
        }
    };
    auto stageA = [&](int buf) {
        __bf16* ar = &As[buf][arow * PITCH + ahalf];
        if (MODE == 0) {
            union { __bf16 h[16]; uint4 q[2]; } cv;
            const float* fs = (const float*)fa;
            #pragma unroll
            for (int i = 0; i < 16; i++) cv.h[i] = (__bf16)fs[i];
            ((uint4*)ar)[0] = cv.q[0]; ((uint4*)ar)[1] = cv.q[1];
        } else {
            ((uint4*)ar)[0] = ua[0]; ((uint4*)ar)[1] = ua[1];
        }
    };
    auto fetchB = [&](int kc) {
        const uint4* pb = (const uint4*)(Wt + (size_t)(n0 + arow) * K + kc * 32 + ahalf);
        ub[0] = pb[0]; ub[1] = pb[1];
    };
    auto stageB = [&](int buf) {
        __bf16* br = &Bs[buf][arow * PITCH + ahalf];
        ((uint4*)br)[0] = ub[0]; ((uint4*)br)[1] = ub[1];
    };
#if HAVE_ASYNC
    auto issueA = [&](int kc, int buf) {  // MODE 1 only
        async_cp32(Abf + (size_t)(m0 + arow) * 1024 + kc * 32 + ahalf,
                   &As[buf][arow * PITCH + ahalf]);
    };
    auto issueB = [&](int kc, int buf) {
        async_cp32(Wt + (size_t)(n0 + arow) * K + kc * 32 + ahalf,
                   &Bs[buf][arow * PITCH + ahalf]);
    };
#endif

    v8f acc[4][2] = {};

    auto compute = [&](int buf) {
        v16bf a[4], b[2];
        #pragma unroll
        for (int mt = 0; mt < 4; mt++)
            a[mt] = load_frag(&As[buf][0], rowbase + mt * 16, idx, half);
        #pragma unroll
        for (int nt = 0; nt < 2; nt++)
            b[nt] = load_frag(&Bs[buf][0], colbase + nt * 16, idx, half);
        #pragma unroll
        for (int mt = 0; mt < 4; mt++)
            #pragma unroll
            for (int nt = 0; nt < 2; nt++)
                acc[mt][nt] = __builtin_amdgcn_wmma_f32_16x16x32_bf16(
                    false, a[mt], false, b[nt], (short)0, acc[mt][nt], false, false);
    };

#if HAVE_ASYNC
    issueB(0, 0);
    if (MODE == 0) { fetchA(0); stageA(0); }
    else           { issueA(0, 0); }
    wait_async0();
    __syncthreads();
    #pragma unroll 1
    for (int kc = 0; kc < NC; ++kc) {
        const int cur = kc & 1;
        if (kc + 1 < NC) {
            issueB(kc + 1, cur ^ 1);
            if (MODE == 0) fetchA(kc + 1); else issueA(kc + 1, cur ^ 1);
        }
        compute(cur);
        if (kc + 1 < NC) {
            if (MODE == 0) stageA(cur ^ 1);
            wait_async0();
        }
        __syncthreads();
    }
#else
    fetchA(0); fetchB(0);
    stageA(0); stageB(0);
    __syncthreads();
    #pragma unroll 1
    for (int kc = 0; kc < NC; ++kc) {
        const int cur = kc & 1;
        if (kc + 1 < NC) { fetchA(kc + 1); fetchB(kc + 1); }
        compute(cur);
        if (kc + 1 < NC) { stageA(cur ^ 1); stageB(cur ^ 1); }
        __syncthreads();
    }
#endif

    #pragma unroll
    for (int nt = 0; nt < 2; nt++) {
        const int col = n0 + colbase + nt * 16 + idx;
        const float bv = bias[col];
        #pragma unroll
        for (int mt = 0; mt < 4; mt++) {
            #pragma unroll
            for (int r = 0; r < 8; r++) {
                const int row = m0 + rowbase + mt * 16 + half * 8 + r;
                const float v = acc[mt][nt][r] + bv;
                Out[(size_t)row * 1024 + col] = (__bf16)lecun_act(v);
            }
        }
    }
}

// ---------------------------------------------------------------------------
// Heads: four GEMMs [M,1024]x[1024,512] sharing each A tile, fused gate.
// Wth[h] = head weights TRANSPOSED [512,1024] bf16.
// Workgroup tile 128x64, 8 waves (4x2), wave tile 32x32 per head.
// ---------------------------------------------------------------------------
__global__ __launch_bounds__(256) void heads_kernel(
    const __bf16* __restrict__ Abf,
    const __bf16* __restrict__ Wt1, const __bf16* __restrict__ Wt2,
    const __bf16* __restrict__ Wt3, const __bf16* __restrict__ Wt4,
    const float* __restrict__ bf1, const float* __restrict__ bf2,
    const float* __restrict__ bta, const float* __restrict__ btb,
    const float* __restrict__ ts,
    float* __restrict__ Out)
{
    constexpr int NC = 32;  // K = 1024

    __shared__ __bf16 As[2][128 * PITCH];
    __shared__ __bf16 Bs[2][4][64 * PITCH];

    const int tid  = threadIdx.x;
    const int lane = tid & 31;
    const int wid  = tid >> 5;
    const int idx  = lane & 15;
    const int half = lane >> 4;

    const int m0 = blockIdx.y * 128;
    const int n0 = blockIdx.x * 64;
    const int rowbase = (wid >> 1) * 32;  // 0..96
    const int colbase = (wid & 1) * 32;   // 0 / 32

    const int arow  = tid >> 1;           // 0..127
    const int ahalf = (tid & 1) * 16;     // 0 / 16
    const int hh    = tid >> 6;           // 0..3 (head this thread stages)
    const int hrow  = tid & 63;           // 0..63 (Bt row within head tile)

    const __bf16* Wh = (hh == 0) ? Wt1 : (hh == 1) ? Wt2 : (hh == 2) ? Wt3 : Wt4;

    uint4 ua[2];
    uint4 ub[4];

    auto fetchA = [&](int kc) {
        const uint4* p4 = (const uint4*)(Abf + (size_t)(m0 + arow) * 1024 + kc * 32 + ahalf);
        ua[0] = p4[0]; ua[1] = p4[1];
    };
    auto stageA = [&](int buf) {
        __bf16* ar = &As[buf][arow * PITCH + ahalf];
        ((uint4*)ar)[0] = ua[0]; ((uint4*)ar)[1] = ua[1];
    };
    auto fetchB = [&](int kc) {
        const uint4* pb = (const uint4*)(Wh + (size_t)(n0 + hrow) * 1024 + kc * 32);
        ub[0] = pb[0]; ub[1] = pb[1]; ub[2] = pb[2]; ub[3] = pb[3];
    };
    auto stageB = [&](int buf) {
        __bf16* br = &Bs[buf][hh][hrow * PITCH];
        ((uint4*)br)[0] = ub[0]; ((uint4*)br)[1] = ub[1];
        ((uint4*)br)[2] = ub[2]; ((uint4*)br)[3] = ub[3];
    };
#if HAVE_ASYNC
    auto issueA = [&](int kc, int buf) {
        async_cp32(Abf + (size_t)(m0 + arow) * 1024 + kc * 32 + ahalf,
                   &As[buf][arow * PITCH + ahalf]);
    };
    auto issueB = [&](int kc, int buf) {
        const __bf16* g = Wh + (size_t)(n0 + hrow) * 1024 + kc * 32;
        __bf16* l = &Bs[buf][hh][hrow * PITCH];
        async_cp32(g, l);
        async_cp32(g + 16, l + 16);
    };
#endif

    v8f acc[4][2][2] = {};  // [head][mt][nt]

    auto compute = [&](int buf) {
        v16bf a[2];
        #pragma unroll
        for (int mt = 0; mt < 2; mt++)
            a[mt] = load_frag(&As[buf][0], rowbase + mt * 16, idx, half);
        #pragma unroll
        for (int h = 0; h < 4; h++) {
            #pragma unroll
            for (int nt = 0; nt < 2; nt++) {
                v16bf b = load_frag(&Bs[buf][h][0], colbase + nt * 16, idx, half);
                #pragma unroll
                for (int mt = 0; mt < 2; mt++)
                    acc[h][mt][nt] = __builtin_amdgcn_wmma_f32_16x16x32_bf16(
                        false, a[mt], false, b, (short)0, acc[h][mt][nt], false, false);
            }
        }
    };

#if HAVE_ASYNC
    issueA(0, 0);
    issueB(0, 0);
    wait_async0();
    __syncthreads();
    #pragma unroll 1
    for (int kc = 0; kc < NC; ++kc) {
        const int cur = kc & 1;
        if (kc + 1 < NC) { issueA(kc + 1, cur ^ 1); issueB(kc + 1, cur ^ 1); }
        compute(cur);
        if (kc + 1 < NC) wait_async0();
        __syncthreads();
    }
#else
    fetchA(0); fetchB(0);
    stageA(0); stageB(0);
    __syncthreads();
    #pragma unroll 1
    for (int kc = 0; kc < NC; ++kc) {
        const int cur = kc & 1;
        if (kc + 1 < NC) { fetchA(kc + 1); fetchB(kc + 1); }
        compute(cur);
        if (kc + 1 < NC) { stageA(cur ^ 1); stageB(cur ^ 1); }
        __syncthreads();
    }
#endif

    #pragma unroll
    for (int nt = 0; nt < 2; nt++) {
        const int col = n0 + colbase + nt * 16 + idx;
        const float b1v = bf1[col], b2v = bf2[col], bav = bta[col], bbv = btb[col];
        #pragma unroll
        for (int mt = 0; mt < 2; mt++) {
            #pragma unroll
            for (int r = 0; r < 8; r++) {
                const int row = m0 + rowbase + mt * 16 + half * 8 + r;
                const float tsv = ts[row];
                const float f1 = fast_tanh(acc[0][mt][nt][r] + b1v);
                const float f2 = fast_tanh(acc[1][mt][nt][r] + b2v);
                const float ta = acc[2][mt][nt][r] + bav;
                const float tb = acc[3][mt][nt][r] + bbv;
                const float s  = fast_sigmoid(ta * tsv + tb);
                Out[(size_t)row * 512 + col] = f1 + s * (f2 - f1);
            }
        }
    }
}

// ---------------------------------------------------------------------------
extern "C" void kernel_launch(void* const* d_in, const int* in_sizes, int n_in,
                              void* d_out, int out_size, void* d_ws, size_t ws_size,
                              hipStream_t stream) {
    const float* input = (const float*)d_in[0];
    const float* hx    = (const float*)d_in[1];
    const float* ts    = (const float*)d_in[2];
    const float* W0    = (const float*)d_in[3];
    const float* b0    = (const float*)d_in[4];
    const float* W1    = (const float*)d_in[5];
    const float* b1    = (const float*)d_in[6];
    const float* Wff1  = (const float*)d_in[7];
    const float* bff1  = (const float*)d_in[8];
    const float* Wff2  = (const float*)d_in[9];
    const float* bff2  = (const float*)d_in[10];
    const float* Wta   = (const float*)d_in[11];
    const float* bta   = (const float*)d_in[12];
    const float* Wtb   = (const float*)d_in[13];
    const float* btb   = (const float*)d_in[14];

    char* ws = (char*)d_ws;
    size_t o = 0;
    __bf16* W0t   = (__bf16*)(ws + o); o += (size_t)768  * 1024 * 2;  // [1024,768]
    __bf16* W1t   = (__bf16*)(ws + o); o += (size_t)1024 * 1024 * 2;  // [1024,1024]
    __bf16* Wff1t = (__bf16*)(ws + o); o += (size_t)1024 * 512  * 2;  // [512,1024]
    __bf16* Wff2t = (__bf16*)(ws + o); o += (size_t)1024 * 512  * 2;
    __bf16* Wtat  = (__bf16*)(ws + o); o += (size_t)1024 * 512  * 2;
    __bf16* Wtbt  = (__bf16*)(ws + o); o += (size_t)1024 * 512  * 2;
    __bf16* H1    = (__bf16*)(ws + o); o += (size_t)M_TOTAL * 1024 * 2;
    __bf16* H2    = (__bf16*)(ws + o); o += (size_t)M_TOTAL * 1024 * 2;

    dim3 blk(256);
    cvt_transpose_bf16<<<dim3(4, 768),  blk, 0, stream>>>(W0,   W0t,   768,  1024);
    cvt_transpose_bf16<<<dim3(4, 1024), blk, 0, stream>>>(W1,   W1t,   1024, 1024);
    cvt_transpose_bf16<<<dim3(2, 1024), blk, 0, stream>>>(Wff1, Wff1t, 1024, 512);
    cvt_transpose_bf16<<<dim3(2, 1024), blk, 0, stream>>>(Wff2, Wff2t, 1024, 512);
    cvt_transpose_bf16<<<dim3(2, 1024), blk, 0, stream>>>(Wta,  Wtat,  1024, 512);
    cvt_transpose_bf16<<<dim3(2, 1024), blk, 0, stream>>>(Wtb,  Wtbt,  1024, 512);

    gemm_backbone<0><<<dim3(8, M_TOTAL / 128), blk, 0, stream>>>(
        input, hx, nullptr, W0t, b0, H1);
    gemm_backbone<1><<<dim3(8, M_TOTAL / 128), blk, 0, stream>>>(
        nullptr, nullptr, H1, W1t, b1, H2);
    heads_kernel<<<dim3(8, M_TOTAL / 128), blk, 0, stream>>>(
        H2, Wff1t, Wff2t, Wtat, Wtbt, bff1, bff2, bta, btb, ts, (float*)d_out);
}
